// AttnRNN_66554813218999
// MI455X (gfx1250) — matrix operand order repaired and checked by reading, not compiled, not run on hardware
//
#include <hip/hip_runtime.h>
#include <hip/hip_bf16.h>
#include <math.h>

// ---------------------------------------------------------------------------
// AttnRNN cell for MI455X (gfx1250): bf16 WMMA GEMMs, double-buffered LDS.
// Dimensions (fixed by the reference): B=4096, D=1024, H=1024, K=8, A=8.
// ---------------------------------------------------------------------------

typedef __attribute__((ext_vector_type(16))) __bf16 v16bf;
typedef __attribute__((ext_vector_type(8)))  float  v8f;

#define BM 128
#define BN 128
#define BK 32
#define APAD 40   // LDS row stride (elements) for As  -> 80B rows, 16B aligned
#define BPAD 40   // LDS row stride (elements) for BsT -> 80B rows, 16B aligned

union Frag16 { uint4 q[2]; v16bf v; };

__device__ __forceinline__ float sigm(float x) { return 1.0f / (1.0f + expf(-x)); }

// ---------------------------------------------------------------------------
// Tiled GEMM: C[M,N] = A1[M,K1]@W1[K1,N] (+ A2[M,K2]@W2[K2,N]) + bias1 + bias2
//   * fp32 operands converted to bf16 while staging in LDS
//   * HAS_MUL: elementwise multiplier on A1 loads (hiddens * i_gt)
//   * MODE 0: store fp32      MODE 1: store sigmoid(fp32)   MODE 3: store bf16
//   * blockIdx.z batches A1/W1/bias1/outBF by the *Batch strides (hs GEMMs)
//   * double-buffered LDS: stage tile k+1 while WMMAs consume tile k
// ---------------------------------------------------------------------------
template <int MODE, bool HAS_A2, bool HAS_MUL>
__global__ __launch_bounds__(256)
void wmma_gemm_kernel(const float* __restrict__ A1, int lda1, int K1,
                      const float* __restrict__ A2, int lda2, int K2,
                      const float* __restrict__ W1, const float* __restrict__ W2,
                      const float* __restrict__ mulA,
                      const float* __restrict__ bias1, const float* __restrict__ bias2,
                      float* __restrict__ outF, __hip_bfloat16* __restrict__ outBF,
                      int M, int N,
                      long long aBatch, long long wBatch, long long biasBatch,
                      long long outBatch)
{
    const int k3 = blockIdx.z;
    const float* A1p = A1 + (size_t)k3 * aBatch;
    const float* W1p = W1 + (size_t)k3 * wBatch;
    const float* b1p = bias1 + (size_t)k3 * biasBatch;

    const int tid  = threadIdx.x;
    const int bm   = blockIdx.y * BM;
    const int bn   = blockIdx.x * BN;
    const int wave = tid >> 5;
    const int lane = tid & 31;
    const int wm   = wave & 3;   // 4 wave-rows  * 32 rows = 128
    const int wn   = wave >> 2;  // 2 wave-cols  * 64 cols = 128
    const int lhalf = lane >> 4; // 0/1
    const int l16   = lane & 15;

    __shared__ __bf16 As [2][BM * APAD];  // [buf][row][k]   row-major in K
    __shared__ __bf16 BsT[2][BN * BPAD];  // [buf][col][k]   transposed: contiguous K

    v8f zero = {};
    v8f acc[2][4];
#pragma unroll
    for (int i = 0; i < 2; ++i)
#pragma unroll
        for (int j = 0; j < 4; ++j) acc[i][j] = zero;

    // global->LDS mapping (256 threads)
    const int arow  = tid >> 1;          // 0..127
    const int acol0 = (tid & 1) * 16;    // 0 / 16
    const int wrow  = tid >> 3;          // 0..31
    const int wcol0 = (tid & 7) * 16;    // 0..112

    // per-thread base pointers (hoisted 64-bit address math)
    const float* aBase1 = A1p + (size_t)(bm + arow) * lda1 + acol0;
    const float* aBase2 = HAS_A2 ? (A2 + (size_t)(bm + arow) * lda2 + acol0) : nullptr;
    const float* mBase  = HAS_MUL ? (mulA + (size_t)(bm + arow) * lda1 + acol0) : nullptr;
    const float* wBase1 = W1p + (size_t)wrow * N + bn + wcol0;
    const float* wBase2 = HAS_A2 ? (W2 + (size_t)wrow * N + bn + wcol0) : nullptr;

    const int totalK = HAS_A2 ? (K1 + K2) : K1;
    const int nkb = totalK / BK;

    // stage one BK-slice (global fp32 -> bf16 LDS) into buffer `buf`
    auto stage = [&](int kbase, int buf) {
        const float* ag; const float* wg; const float* mg = nullptr;
        if (!HAS_A2 || kbase < K1) {
            ag = aBase1 + kbase;
            wg = wBase1 + (size_t)kbase * N;
            if (HAS_MUL) mg = mBase + kbase;
        } else {
            const int kloc = kbase - K1;
            ag = aBase2 + kloc;
            wg = wBase2 + (size_t)kloc * N;
        }
        // ---- A tile: 16 consecutive K-values for this thread's row ----
        {
            const float4* g4 = (const float4*)ag;
            __bf16* dst = &As[buf][arow * APAD + acol0];
#pragma unroll
            for (int j = 0; j < 4; ++j) {
                float4 v = g4[j];
                if (HAS_MUL) {
                    const float4 m = ((const float4*)mg)[j];
                    v.x *= m.x; v.y *= m.y; v.z *= m.z; v.w *= m.w;
                }
                dst[j * 4 + 0] = (__bf16)v.x; dst[j * 4 + 1] = (__bf16)v.y;
                dst[j * 4 + 2] = (__bf16)v.z; dst[j * 4 + 3] = (__bf16)v.w;
            }
        }
        // ---- W tile: one K-row, 16 N-values, stored transposed ----
        {
            const float4* g4 = (const float4*)wg;
            __bf16* bs = BsT[buf];
#pragma unroll
            for (int j = 0; j < 4; ++j) {
                float4 v = g4[j];
                const int n0 = wcol0 + j * 4;
                bs[(n0 + 0) * BPAD + wrow] = (__bf16)v.x;
                bs[(n0 + 1) * BPAD + wrow] = (__bf16)v.y;
                bs[(n0 + 2) * BPAD + wrow] = (__bf16)v.z;
                bs[(n0 + 3) * BPAD + wrow] = (__bf16)v.w;
            }
        }
        // prefetch the slice after this one (emits global_prefetch_b8)
        const int kn = kbase + BK;
        if (kn < totalK) {
            const float* pa; const float* pw;
            if (!HAS_A2 || kn < K1) { pa = aBase1 + kn; pw = wBase1 + (size_t)kn * N; }
            else { const int kl = kn - K1; pa = aBase2 + kl; pw = wBase2 + (size_t)kl * N; }
            __builtin_prefetch(pa, 0, 1);
            __builtin_prefetch(pw, 0, 1);
        }
    };

    stage(0, 0);
    int cur = 0;

    for (int kb = 0; kb < nkb; ++kb) {
        __syncthreads();   // buf[cur] staged & all waves done with buf[cur^1]
        if (kb + 1 < nkb) stage((kb + 1) * BK, cur ^ 1);

        // ---- fragments per ISA 7.12.2 layouts (from buf[cur]) ----
        Frag16 afr[2], bfr[4];
#pragma unroll
        for (int mt = 0; mt < 2; ++mt) {
            const __bf16* row = &As[cur][(wm * 32 + mt * 16 + l16) * APAD];
            afr[mt].q[0] = *(const uint4*)(row + lhalf * 8);        // K = 8*half + 0..7
            afr[mt].q[1] = *(const uint4*)(row + 16 + lhalf * 8);   // K = 16 + 8*half + 0..7
        }
#pragma unroll
        for (int nt = 0; nt < 4; ++nt) {
            const __bf16* row = &BsT[cur][(wn * 64 + nt * 16 + l16) * BPAD];
            bfr[nt].q[0] = *(const uint4*)(row + lhalf * 8);        // K = 8*half + 0..7
            bfr[nt].q[1] = *(const uint4*)(row + 16 + lhalf * 8);   // K = 16 + 8*half + 0..7
        }
#pragma unroll
        for (int mt = 0; mt < 2; ++mt)
#pragma unroll
            for (int nt = 0; nt < 4; ++nt)
                acc[mt][nt] = __builtin_amdgcn_wmma_f32_16x16x32_bf16(
                    false, afr[mt].v, false, bfr[nt].v, (short)0, acc[mt][nt],
                    false, false);

        cur ^= 1;
    }

    // ---- epilogue (C/D layout: VGPR r -> M = 8*half + r, N = lane%16) ----
#pragma unroll
    for (int mt = 0; mt < 2; ++mt) {
#pragma unroll
        for (int nt = 0; nt < 4; ++nt) {
            const int col = bn + wn * 64 + nt * 16 + l16;
            const float bb = b1p[col] + (HAS_A2 ? bias2[col] : 0.0f);
#pragma unroll
            for (int r = 0; r < 8; ++r) {
                const int row = bm + wm * 32 + mt * 16 + lhalf * 8 + r;
                float v = acc[mt][nt][r] + bb;
                if (MODE == 1) v = sigm(v);
                if (MODE == 3) {
                    outBF[(size_t)k3 * outBatch + (size_t)row * N + col] = __float2bfloat16(v);
                } else {
                    outF[(size_t)row * N + col] = v;
                }
            }
        }
    }
}

// ---------------------------------------------------------------------------
// Additive attention: one wave per batch row b.
//   uv[b,k] = tanh(hs[k,b,:]@attnW + attnb) . attnWu  ; alphas = softmax_k(uv)
// ---------------------------------------------------------------------------
__global__ __launch_bounds__(256)
void attn_kernel(const __hip_bfloat16* __restrict__ hs,   // [K,B,H] bf16
                 const float* __restrict__ attnW,         // [H,8]
                 const float* __restrict__ attnb,         // [8]
                 const float* __restrict__ attnWu,        // [8]
                 float* __restrict__ alphas,              // [B,8]
                 int Bsz, int H)
{
    const int wave = threadIdx.x >> 5;
    const int lane = threadIdx.x & 31;
    const int b = blockIdx.x * 8 + wave;
    if (b >= Bsz) return;

    float uv[8];
#pragma unroll
    for (int k = 0; k < 8; ++k) {
        float s0=0,s1=0,s2=0,s3=0,s4=0,s5=0,s6=0,s7=0;
        const __hip_bfloat16* hrow = hs + ((size_t)k * Bsz + b) * H;
        for (int h = lane; h < H; h += 32) {
            const float v = __bfloat162float(hrow[h]);
            const float4* w4 = (const float4*)(attnW + (size_t)h * 8);
            const float4 wa = w4[0], wb = w4[1];
            s0 += v * wa.x; s1 += v * wa.y; s2 += v * wa.z; s3 += v * wa.w;
            s4 += v * wb.x; s5 += v * wb.y; s6 += v * wb.z; s7 += v * wb.w;
        }
        float s[8] = {s0,s1,s2,s3,s4,s5,s6,s7};
#pragma unroll
        for (int a = 0; a < 8; ++a) {
            float t = s[a];
#pragma unroll
            for (int off = 16; off > 0; off >>= 1) t += __shfl_xor(t, off, 32);
            s[a] = t;
        }
        float dot = 0.0f;
#pragma unroll
        for (int a = 0; a < 8; ++a) dot += tanhf(s[a] + attnb[a]) * attnWu[a];
        uv[k] = dot;   // identical across lanes after butterfly reduce
    }
    float mx = uv[0];
#pragma unroll
    for (int k = 1; k < 8; ++k) mx = fmaxf(mx, uv[k]);
    float ssum = 0.0f, e[8];
#pragma unroll
    for (int k = 0; k < 8; ++k) { e[k] = expf(uv[k] - mx); ssum += e[k]; }
    if (lane == 0) {
        const float inv = 1.0f / ssum;
#pragma unroll
        for (int k = 0; k < 8; ++k) alphas[(size_t)b * 8 + k] = e[k] * inv;
    }
}

// ---------------------------------------------------------------------------
// Final elementwise: u = u_pre + sum_k alphas*hs ; LSTM-style update.
// ---------------------------------------------------------------------------
__global__ __launch_bounds__(256)
void final_kernel(const float* __restrict__ f_pre, const float* __restrict__ o_pre,
                  const float* __restrict__ u_pre, const float* __restrict__ cells,
                  const __hip_bfloat16* __restrict__ hs, const float* __restrict__ alphas,
                  float* __restrict__ out_h, float* __restrict__ out_c,
                  int Bsz, int H)
{
    const size_t total = (size_t)Bsz * H;
    const size_t idx = (size_t)blockIdx.x * blockDim.x + threadIdx.x;
    if (idx >= total) return;
    const int b = (int)(idx / H);
    const float* al = alphas + (size_t)b * 8;

    float u = u_pre[idx];
#pragma unroll
    for (int k = 0; k < 8; ++k)
        u += al[k] * __bfloat162float(hs[(size_t)k * total + idx]);

    const float fs = sigm(f_pre[idx]);
    const float c_last = cells[(size_t)7 * total + idx];
    const float cn = c_last * fs + tanhf(u) * (1.0f - fs);
    out_h[idx] = tanhf(cn) * sigm(o_pre[idx]);
    out_c[idx] = cn;
}

// ---------------------------------------------------------------------------
extern "C" void kernel_launch(void* const* d_in, const int* in_sizes, int n_in,
                              void* d_out, int out_size, void* d_ws, size_t ws_size,
                              hipStream_t stream)
{
    const int B = 4096, D = 1024, H = 1024, K = 8;

    const float* x    = (const float*)d_in[0];
    const float* hid  = (const float*)d_in[1];
    const float* cel  = (const float*)d_in[2];
    const float* Wfx  = (const float*)d_in[3];  const float* bfx = (const float*)d_in[4];
    const float* Wox  = (const float*)d_in[5];  const float* box_= (const float*)d_in[6];
    const float* Wix  = (const float*)d_in[7];  const float* bix = (const float*)d_in[8];
    const float* Wux  = (const float*)d_in[9];  const float* bux = (const float*)d_in[10];
    const float* Wfh  = (const float*)d_in[11]; const float* bfh = (const float*)d_in[12];
    const float* Woh  = (const float*)d_in[13]; const float* boh = (const float*)d_in[14];
    const float* Wih  = (const float*)d_in[15]; const float* bih = (const float*)d_in[16];
    const float* Wk   = (const float*)d_in[17]; const float* bk  = (const float*)d_in[18];
    const float* attnW  = (const float*)d_in[19];
    const float* attnb  = (const float*)d_in[20];
    const float* attnWu = (const float*)d_in[21];

    const float* h_last = hid + (size_t)(K - 1) * B * H;

    // workspace carve-out (~129 MB)
    char* ws = (char*)d_ws;
    float* f_pre = (float*)ws;                ws += (size_t)B * H * sizeof(float);
    float* o_pre = (float*)ws;                ws += (size_t)B * H * sizeof(float);
    float* igt   = (float*)ws;                ws += (size_t)B * H * sizeof(float);
    float* u_pre = (float*)ws;                ws += (size_t)B * H * sizeof(float);
    __hip_bfloat16* hs = (__hip_bfloat16*)ws; ws += (size_t)K * B * H * sizeof(__hip_bfloat16);
    float* alphas = (float*)ws;               ws += (size_t)B * K * sizeof(float);

    dim3 blk(256);
    dim3 grdG(H / BN, B / BM, 1);    // (8, 32)
    dim3 grdH(H / BN, B / BM, K);    // (8, 32, 8)

    // 1) i gate (sigmoid epilogue) -> i_gt  [must precede hs GEMMs]
    wmma_gemm_kernel<1, true, false><<<grdG, blk, 0, stream>>>(
        x, D, D, h_last, H, H, Wix, Wih, nullptr, bix, bih,
        igt, nullptr, B, H, 0, 0, 0, 0);
    // 2) f, o gates and u_pre
    wmma_gemm_kernel<0, true, false><<<grdG, blk, 0, stream>>>(
        x, D, D, h_last, H, H, Wfx, Wfh, nullptr, bfx, bfh,
        f_pre, nullptr, B, H, 0, 0, 0, 0);
    wmma_gemm_kernel<0, true, false><<<grdG, blk, 0, stream>>>(
        x, D, D, h_last, H, H, Wox, Woh, nullptr, box_, boh,
        o_pre, nullptr, B, H, 0, 0, 0, 0);
    wmma_gemm_kernel<0, false, false><<<grdG, blk, 0, stream>>>(
        x, D, D, nullptr, 0, 0, Wux, nullptr, nullptr, bux, nullptr,
        u_pre, nullptr, B, H, 0, 0, 0, 0);
    // 3) hs[k] = (hiddens[k] * i_gt) @ Wk[k] + bk[k]   (batched over z = k)
    wmma_gemm_kernel<3, false, true><<<grdH, blk, 0, stream>>>(
        hid, H, H, nullptr, 0, 0, Wk, nullptr, igt, bk, nullptr,
        nullptr, hs, B, H,
        (long long)B * H, (long long)H * H, (long long)H, (long long)B * H);
    // 4) additive attention -> alphas
    attn_kernel<<<dim3(B / 8), blk, 0, stream>>>(hs, attnW, attnb, attnWu, alphas, B, H);
    // 5) final elementwise update -> d_out = [hidden_next | cell_next]
    float* out_h = (float*)d_out;
    float* out_c = out_h + (size_t)B * H;
    const size_t total = (size_t)B * H;
    final_kernel<<<dim3((unsigned)((total + 255) / 256)), blk, 0, stream>>>(
        f_pre, o_pre, u_pre, cel, hs, alphas, out_h, out_c, B, H);
}
